// OneHeaderAttention_43928925504077
// MI455X (gfx1250) — matrix operand order repaired and checked by reading, not compiled, not run on hardware
//
#include <hip/hip_runtime.h>

// Problem constants (match reference)
#define Hh 1024
#define BB 4
#define SQ 2048
#define SKk 2048

// gfx1250 async global->LDS staging (GLOBAL_LOAD_ASYNC_TO_LDS_B128).
#define USE_ASYNC_COPY 1

typedef __attribute__((ext_vector_type(16))) __bf16 v16bf;
typedef __attribute__((ext_vector_type(8)))  float  v8f;

__device__ __forceinline__ unsigned short f2bf(float f) {
  unsigned u = __builtin_bit_cast(unsigned, f);
  u += 0x7FFFu + ((u >> 16) & 1u);          // round-to-nearest-even
  return (unsigned short)(u >> 16);
}

#if USE_ASYNC_COPY
// LDS generic pointers carry the LDS byte offset in addr[31:0] (ISA 10.2).
__device__ __forceinline__ void async_ld_b128(unsigned lds_off,
                                              const void* gptr) {
  asm volatile("global_load_async_to_lds_b128 %0, %1, off"
               :: "v"(lds_off), "v"((unsigned long long)gptr)
               : "memory");
}
#endif

// ---------------------------------------------------------------------------
// fp32 -> bf16 conversion (grid-stride)
// ---------------------------------------------------------------------------
__global__ __launch_bounds__(256) void cvt_f32_to_bf16(
    const float* __restrict__ src, unsigned short* __restrict__ dst, size_t n) {
  size_t i = (size_t)blockIdx.x * blockDim.x + threadIdx.x;
  size_t stride = (size_t)gridDim.x * blockDim.x;
  for (; i < n; i += stride) dst[i] = f2bf(src[i]);
}

// ---------------------------------------------------------------------------
// Tiled WMMA GEMM, C = scale * (A @ B^T) + bias.
// A: [M x K] bf16 row-major (lda), Bm: [N x K] bf16 row-major (ldb).
// OUTMODE 0: f32 out [M x N]; 1: bf16 out; 2: bf16 out transposed
// (out[n*ldc + m]).  Batched via blockIdx.z with strides sA/sB/sC.
//
// Block = 128 threads = 4 waves in a 2(M) x 2(N) grid.
// Block tile 64M x 128N, K-chunk 32; wave tile 32M x 64N = 8 WMMA / K-step.
// LDS double-buffered; staged with async global->LDS B128 loads.
// __launch_bounds__(128, 1): relax VGPR budget so all fragments stay live
// in disjoint registers -> back-to-back v_wmma without hazard NOPs.
// ---------------------------------------------------------------------------
template <int OUTMODE>
__global__ __launch_bounds__(128, 1) void gemm_nt(
    const unsigned short* __restrict__ A, int lda, long long sA,
    const unsigned short* __restrict__ Bm, int ldb, long long sB,
    const float* __restrict__ bias, void* __restrict__ Cp, int ldc,
    long long sC, int M, int N, int Kdim, float scale) {
  // row stride 17 u32 (68B) to spread the 64 LDS banks
  __shared__ unsigned As[2][64 * 17];    // 64 rows  x 32 bf16
  __shared__ unsigned Bs[2][128 * 17];   // 128 rows x 32 bf16

  const int tid   = threadIdx.x;      // 0..127
  const int wave  = tid >> 5;         // 0..3
  const int lane  = tid & 31;
  const int laneN = lane & 15;
  const int hi    = lane >> 4;        // lane-half select
  const int wm    = wave >> 1;        // wave M index (0..1)
  const int wn    = wave & 1;         // wave N index (0..1)

  const int m0 = blockIdx.x * 64;
  const int n0 = blockIdx.y * 128;
  const int b  = blockIdx.z;

  const unsigned short* Ab = A + (size_t)b * sA;
  const unsigned short* Bb = Bm + (size_t)b * sB;

  v8f acc[2][4];
#pragma unroll
  for (int mi = 0; mi < 2; ++mi)
#pragma unroll
    for (int t = 0; t < 4; ++t)
      acc[mi][t] = (v8f){0.f, 0.f, 0.f, 0.f, 0.f, 0.f, 0.f, 0.f};

  // ---- staging: 6 x B128 per thread per K-step (2 for A, 4 for B) ----
  auto stage = [&](int buf, int k0) {
#pragma unroll
    for (int it = 0; it < 2; ++it) {            // A: 64 rows x 4 quarters
      const int i = tid + it * 128;
      const int row = i >> 2, q = i & 3;
      const unsigned short* g = Ab + (size_t)(m0 + row) * lda + k0 + q * 8;
#if USE_ASYNC_COPY
      async_ld_b128((unsigned)(size_t)&As[buf][row * 17 + q * 4], g);
#else
      uint4 v = *(const uint4*)g;
      unsigned* d = &As[buf][row * 17 + q * 4];
      d[0] = v.x; d[1] = v.y; d[2] = v.z; d[3] = v.w;
#endif
    }
#pragma unroll
    for (int it = 0; it < 4; ++it) {            // B: 128 rows x 4 quarters
      const int i = tid + it * 128;
      const int row = i >> 2, q = i & 3;
      const unsigned short* g = Bb + (size_t)(n0 + row) * ldb + k0 + q * 8;
#if USE_ASYNC_COPY
      async_ld_b128((unsigned)(size_t)&Bs[buf][row * 17 + q * 4], g);
#else
      uint4 v = *(const uint4*)g;
      unsigned* d = &Bs[buf][row * 17 + q * 4];
      d[0] = v.x; d[1] = v.y; d[2] = v.z; d[3] = v.w;
#endif
    }
  };

  const int nk = Kdim >> 5;
  stage(0, 0);
  for (int kb = 0; kb < nk; ++kb) {
    const int cur = kb & 1;
    if (kb + 1 < nk) {
      stage(cur ^ 1, (kb + 1) * 32);
#if USE_ASYNC_COPY
      // current stage's 6 loads precede the 6 just issued -> <=6 means done
      asm volatile("s_wait_asynccnt 6" ::: "memory");
#endif
    }
#if USE_ASYNC_COPY
    else {
      asm volatile("s_wait_asynccnt 0" ::: "memory");
    }
#endif
    __syncthreads();

    // ---- preload ALL fragments for this K-step into disjoint registers ----
    // A fragment (16x32 bf16): u32 slot r holds K pair; r<4: K=2r(+8 hi),
    // r>=4: K=16+2(r-4)(+8 hi).
    union Frag { v16bf v; unsigned u[8]; };
    Frag af[2], bfr[4];
#pragma unroll
    for (int mi = 0; mi < 2; ++mi) {
      const int arow = wm * 32 + mi * 16 + laneN;
#pragma unroll
      for (int r = 0; r < 8; ++r) {
        const int kk = (r < 4) ? (hi * 4 + r) : (8 + hi * 4 + (r - 4));
        af[mi].u[r] = As[cur][arow * 17 + kk];
      }
    }
#pragma unroll
    for (int t = 0; t < 4; ++t) {
      const int brow = wn * 64 + t * 16 + laneN;
#pragma unroll
      for (int v = 0; v < 8; ++v) bfr[t].u[v] = Bs[cur][brow * 17 + ((v < 4) ? (hi * 4 + v) : (8 + hi * 4 + (v - 4)))];
    }

    // ---- 8 back-to-back WMMAs (disjoint regs: no hazard NOPs needed) ----
#pragma unroll
    for (int t = 0; t < 4; ++t)
#pragma unroll
      for (int mi = 0; mi < 2; ++mi)
        acc[mi][t] = __builtin_amdgcn_wmma_f32_16x16x32_bf16(
            false, af[mi].v, false, bfr[t].v, (short)0, acc[mi][t], false,
            false);

    __syncthreads();   // protect cur buffer before it is re-staged
  }

  // Epilogue. C/D layout: VGPR j -> row (hi*8 + j), col laneN.
#pragma unroll
  for (int mi = 0; mi < 2; ++mi) {
#pragma unroll
    for (int t = 0; t < 4; ++t) {
      const int gn = n0 + wn * 64 + t * 16 + laneN;
      const float bv = bias ? bias[gn] : 0.f;
#pragma unroll
      for (int j = 0; j < 8; ++j) {
        const int gm = m0 + wm * 32 + mi * 16 + hi * 8 + j;
        const float v = acc[mi][t][j] * scale + bv;
        if (OUTMODE == 0)
          ((float*)Cp)[(size_t)b * sC + (size_t)gm * ldc + gn] = v;
        else if (OUTMODE == 1)
          ((unsigned short*)Cp)[(size_t)b * sC + (size_t)gm * ldc + gn] =
              f2bf(v);
        else
          ((unsigned short*)Cp)[(size_t)b * sC + (size_t)gn * ldc + gm] =
              f2bf(v);
      }
    }
  }
}

// ---------------------------------------------------------------------------
// Row softmax: f32 scores [B*SQ, SK] -> bf16 probabilities, with key mask.
// ---------------------------------------------------------------------------
__global__ __launch_bounds__(256) void softmax_rows(
    const float* __restrict__ S, unsigned short* __restrict__ P,
    const unsigned char* __restrict__ mask) {
  __shared__ float red[256];
  const int r = blockIdx.x;
  const int b = r / SQ;
  const float* srow = S + (size_t)r * SKk;
  const unsigned char* mrow = mask + (size_t)b * SKk;
  const int tid = threadIdx.x;

  float mx = -3.0e38f;
  for (int i = tid; i < SKk; i += 256) {
    const float s = mrow[i] ? srow[i] : -3.0e38f;
    mx = fmaxf(mx, s);
  }
  red[tid] = mx;
  __syncthreads();
  for (int o = 128; o > 0; o >>= 1) {
    if (tid < o) red[tid] = fmaxf(red[tid], red[tid + o]);
    __syncthreads();
  }
  mx = red[0];
  __syncthreads();

  float sum = 0.f;
  for (int i = tid; i < SKk; i += 256)
    sum += mrow[i] ? __expf(srow[i] - mx) : 0.f;
  red[tid] = sum;
  __syncthreads();
  for (int o = 128; o > 0; o >>= 1) {
    if (tid < o) red[tid] += red[tid + o];
    __syncthreads();
  }
  const float inv = 1.0f / red[0];

  for (int i = tid; i < SKk; i += 256) {
    const float p = mrow[i] ? __expf(srow[i] - mx) * inv : 0.f;
    P[(size_t)r * SKk + i] = f2bf(p);
  }
}

// ---------------------------------------------------------------------------
extern "C" void kernel_launch(void* const* d_in, const int* in_sizes, int n_in,
                              void* d_out, int out_size, void* d_ws,
                              size_t ws_size, hipStream_t stream) {
  const float* inputs = (const float*)d_in[0];
  const float* memory = (const float*)d_in[1];
  const unsigned char* mmask = (const unsigned char*)d_in[2];
  const float* Wq = (const float*)d_in[3];
  const float* bq = (const float*)d_in[4];
  const float* Wk = (const float*)d_in[5];
  const float* bk = (const float*)d_in[6];
  const float* Wv = (const float*)d_in[7];
  const float* bv = (const float*)d_in[8];
  const float* Wo = (const float*)d_in[9];
  const float* bo = (const float*)d_in[10];

  const size_t nx = (size_t)BB * SQ * Hh;
  const size_t nm = (size_t)BB * SKk * Hh;
  const size_t nw = (size_t)Hh * Hh;
  const size_t ns = (size_t)BB * SQ * SKk;

  char* ws = (char*)d_ws;
  size_t off = 0;
  auto alloc = [&](size_t bytes) -> char* {
    char* p = ws + off;
    off += (bytes + 255) & ~(size_t)255;
    return p;
  };
  unsigned short* xb  = (unsigned short*)alloc(nx * 2);
  unsigned short* mb  = (unsigned short*)alloc(nm * 2);
  unsigned short* wqb = (unsigned short*)alloc(nw * 2);
  unsigned short* wkb = (unsigned short*)alloc(nw * 2);
  unsigned short* wvb = (unsigned short*)alloc(nw * 2);
  unsigned short* wob = (unsigned short*)alloc(nw * 2);
  unsigned short* qb  = (unsigned short*)alloc(nx * 2);
  unsigned short* kb  = (unsigned short*)alloc(nm * 2);
  unsigned short* vtb = (unsigned short*)alloc(nm * 2);  // V^T [H][B*SK]
  float*          sf  = (float*)alloc(ns * 4);
  unsigned short* pb  = (unsigned short*)alloc(ns * 2);
  unsigned short* ab  = (unsigned short*)alloc(nx * 2);

  cvt_f32_to_bf16<<<2048, 256, 0, stream>>>(inputs, xb, nx);
  cvt_f32_to_bf16<<<2048, 256, 0, stream>>>(memory, mb, nm);
  cvt_f32_to_bf16<<<512, 256, 0, stream>>>(Wq, wqb, nw);
  cvt_f32_to_bf16<<<512, 256, 0, stream>>>(Wk, wkb, nw);
  cvt_f32_to_bf16<<<512, 256, 0, stream>>>(Wv, wvb, nw);
  cvt_f32_to_bf16<<<512, 256, 0, stream>>>(Wo, wob, nw);

  const int Mrows = BB * SQ;  // 8192
  const dim3 blk(128);

  // Q = x @ Wq^T + bq
  gemm_nt<1><<<dim3(Mrows / 64, Hh / 128, 1), blk, 0, stream>>>(
      xb, Hh, 0, wqb, Hh, 0, bq, qb, Hh, 0, Mrows, Hh, Hh, 1.0f);
  // K = mem @ Wk^T + bk
  gemm_nt<1><<<dim3(Mrows / 64, Hh / 128, 1), blk, 0, stream>>>(
      mb, Hh, 0, wkb, Hh, 0, bk, kb, Hh, 0, Mrows, Hh, Hh, 1.0f);
  // V = mem @ Wv^T + bv, stored transposed: vtb[h][b*SK + s]
  gemm_nt<2><<<dim3(Mrows / 64, Hh / 128, 1), blk, 0, stream>>>(
      mb, Hh, 0, wvb, Hh, 0, bv, vtb, Mrows, 0, Mrows, Hh, Hh, 1.0f);

  // scores = (Q @ K^T) / sqrt(H)
  gemm_nt<0><<<dim3(SQ / 64, SKk / 128, BB), blk, 0, stream>>>(
      qb, Hh, (long long)SQ * Hh, kb, Hh, (long long)SKk * Hh, nullptr, sf,
      SKk, (long long)SQ * SKk, SQ, SKk, Hh, 0.03125f);

  softmax_rows<<<BB * SQ, 256, 0, stream>>>(sf, pb, mmask);

  // attn = P @ V
  gemm_nt<1><<<dim3(SQ / 64, Hh / 128, BB), blk, 0, stream>>>(
      pb, SKk, (long long)SQ * SKk, vtb, Mrows, (long long)SKk, nullptr, ab,
      Hh, (long long)SQ * Hh, SQ, Hh, SKk, 1.0f);

  // out = attn @ Wo^T + bo
  gemm_nt<0><<<dim3(Mrows / 64, Hh / 128, 1), blk, 0, stream>>>(
      ab, Hh, 0, wob, Hh, 0, bo, (float*)d_out, Hh, 0, Mrows, Hh, Hh, 1.0f);
}
